// HeCo_40776419508609
// MI455X (gfx1250) — matrix operand order repaired and checked by reading, not compiled, not run on hardware
//
#include <hip/hip_runtime.h>
#include <hip/hip_bf16.h>
#include <math.h>

// ---------------------------------------------------------------------------
// HeCo contrastive loss on MI455X (gfx1250, wave32, WMMA).
// All dense GEMMs use v_wmma_f32_16x16x32_f16 (f16 inputs, fp32 accum).
// The 8192x8192x64 similarity GEMM is fused with exp + row/col/pos-masked
// reductions. Symmetric tile-pairing: each wave computes sim tiles (tm,tn)
// AND (tn,tm), staging both pos tiles through LDS, so the 268MB pos matrix
// is streamed from HBM exactly once, fully coalesced (dominant HBM term).
// ---------------------------------------------------------------------------

typedef __attribute__((ext_vector_type(16))) _Float16 v16h;
typedef __attribute__((ext_vector_type(8)))  float    v8f;

#define NN     8192
#define NNBR   32768
#define DIN    128
#define DH     64
#define NBSZ   8
#define MPDEG  16
#define INV_TAU 1.25f   // 1/0.8

// ---------------- f32 -> f16 conversion (feats + weights) ------------------
__global__ void heco_cvt_f16(const float* __restrict__ src,
                             _Float16* __restrict__ dst, int n) {
    for (int i = blockIdx.x * blockDim.x + threadIdx.x; i < n;
         i += gridDim.x * blockDim.x)
        dst[i] = (_Float16)src[i];
}

// ---------------- generic WMMA linear: Y = act(X @ W^T + b) ----------------
// X: M x K f16 row-major, W: N x K f16 row-major (so W^T is K x N).
// One wave per 16x16 output tile; K looped in steps of 32.
// act: 0=none, 1=elu, 2=tanh.  Writes Yh (f16) and/or Yf (f32) if non-null.
__global__ void heco_wmma_linear(const _Float16* __restrict__ X,
                                 const _Float16* __restrict__ W,
                                 const float* __restrict__ bias,
                                 _Float16* __restrict__ Yh,
                                 float* __restrict__ Yf,
                                 int M, int N, int K, int act) {
    const int wave = threadIdx.x >> 5;
    const int lane = threadIdx.x & 31;
    const int tilesN = N >> 4;
    const int tile = blockIdx.x * (blockDim.x >> 5) + wave;
    const int totTiles = (M >> 4) * tilesN;
    if (tile >= totTiles) return;            // wave-uniform: EXEC stays all-1s
    const int tm = (tile / tilesN) << 4;
    const int tn = (tile % tilesN) << 4;
    const int l15 = lane & 15;
    const int grp = lane >> 4;

    // A fragment: lane holds row (tm+l15); 16B chunks at k = grp*8, grp*8+16
    const _Float16* xrow = X + (size_t)(tm + l15) * K + grp * 8;
    // B fragment: lane holds column (tn+l15) = row of W; 16B chunks at k = grp*8, grp*8+16
    const _Float16* wrow = W + (size_t)(tn + l15) * K + grp * 8;

    v8f c = {};
    for (int k0 = 0; k0 < K; k0 += 32) {
        union { v16h v; uint4 q[2]; } a, b;
        a.q[0] = *(const uint4*)(xrow + k0);
        a.q[1] = *(const uint4*)(xrow + k0 + 16);
        b.q[0] = *(const uint4*)(wrow + k0);
        b.q[1] = *(const uint4*)(wrow + k0 + 16);
        c = __builtin_amdgcn_wmma_f32_16x16x32_f16(
                false, a.v, false, b.v, (short)0, c, false, false);
    }
    const float bv = bias ? bias[tn + l15] : 0.0f;
#pragma unroll
    for (int v = 0; v < 8; ++v) {
        const int m = tm + v + grp * 8;      // C/D layout: M = v + 8*grp
        const int n = tn + l15;              //             N = lane&15
        float y = c[v] + bv;
        if (act == 1)      y = y > 0.f ? y : (__expf(y) - 1.f);
        else if (act == 2) y = tanhf(y);
        if (Yh) Yh[(size_t)m * N + n] = (_Float16)y;
        if (Yf) Yf[(size_t)m * N + n] = y;
    }
}

// ---------------- GAT over fixed-degree-8 bipartite graph ------------------
__global__ void heco_gat(const _Float16* __restrict__ hs,   // NNBR x 64 (src)
                         const _Float16* __restrict__ hd,   // NN x 64 (dst)
                         const int* __restrict__ src,
                         const float* __restrict__ attn_l,
                         const float* __restrict__ attn_r,
                         _Float16* __restrict__ out) {
    const int i = blockIdx.x, t = threadIdx.x;               // blockDim = 64
    __shared__ float e_sh[NBSZ];
    __shared__ int   s_sh[NBSZ];
    if (t < NBSZ) {
        const int s = src[i * NBSZ + t];
        s_sh[t] = s;
        float el = 0.f, er = 0.f;
        for (int k = 0; k < DH; ++k) {
            el += (float)hs[(size_t)s * DH + k] * attn_l[k];
            er += (float)hd[(size_t)i * DH + k] * attn_r[k];
        }
        const float e = el + er;
        e_sh[t] = e > 0.f ? e : 0.01f * e;                   // leaky_relu
    }
    __syncthreads();
    float m = e_sh[0];
#pragma unroll
    for (int e = 1; e < NBSZ; ++e) m = fmaxf(m, e_sh[e]);
    float a[NBSZ], ssum = 0.f;
#pragma unroll
    for (int e = 0; e < NBSZ; ++e) { a[e] = __expf(e_sh[e] - m); ssum += a[e]; }
    float acc = 0.f;
#pragma unroll
    for (int e = 0; e < NBSZ; ++e)
        acc += (a[e] / ssum) * (float)hs[(size_t)s_sh[e] * DH + t];
    acc = acc > 0.f ? acc : (__expf(acc) - 1.f);             // elu
    out[(size_t)i * DH + t] = (_Float16)acc;
}

// ---------------- GCN(norm='right') aggregation, deg == 16 -----------------
__global__ void heco_gcn_agg(const _Float16* __restrict__ xw,
                             const int* __restrict__ src,
                             const float* __restrict__ bias,
                             const float* __restrict__ alpha_p,
                             _Float16* __restrict__ out) {
    const int i = blockIdx.x, t = threadIdx.x;               // blockDim = 64
    float acc = 0.f;
#pragma unroll
    for (int e = 0; e < MPDEG; ++e)
        acc += (float)xw[(size_t)src[i * MPDEG + e] * DH + t];
    float h = acc * (1.0f / MPDEG) + bias[t];
    const float alpha = alpha_p[0];
    h = h >= 0.f ? h : alpha * h;                            // PReLU
    out[(size_t)i * DH + t] = (_Float16)h;
}

// ------- semantic attention: slot += mean_i sum_t tanh_out[i,t]*vec[t] -----
__global__ void heco_semred(const _Float16* __restrict__ th,
                            const float* __restrict__ vec,
                            float* __restrict__ slot) {
    __shared__ float sd[256];
    float s = 0.f;
    for (int idx = blockIdx.x * blockDim.x + threadIdx.x; idx < NN * DH;
         idx += gridDim.x * blockDim.x)
        s += (float)th[idx] * vec[idx & (DH - 1)];
    sd[threadIdx.x] = s; __syncthreads();
    for (int st = 128; st > 0; st >>= 1) {
        if (threadIdx.x < st) sd[threadIdx.x] += sd[threadIdx.x + st];
        __syncthreads();
    }
    if (threadIdx.x == 0) atomicAdd(slot, sd[0] * (1.0f / NN));
}

__global__ void heco_softmax2(const float* __restrict__ w,
                              float* __restrict__ beta) {
    const float m = fmaxf(w[0], w[1]);
    const float e0 = __expf(w[0] - m), e1 = __expf(w[1] - m);
    beta[0] = e0 / (e0 + e1);
    beta[1] = e1 / (e0 + e1);
}

__global__ void heco_combine2(const _Float16* __restrict__ a,
                              const _Float16* __restrict__ b,
                              const float* __restrict__ beta,
                              _Float16* __restrict__ z, int n) {
    const float b0 = beta[0], b1 = beta[1];
    for (int i = blockIdx.x * blockDim.x + threadIdx.x; i < n;
         i += gridDim.x * blockDim.x)
        z[i] = (_Float16)(b0 * (float)a[i] + b1 * (float)b[i]);
}

// ---------------- L2 row-normalize (f32 in -> f16 out) ---------------------
__global__ void heco_l2norm(const float* __restrict__ x,
                            _Float16* __restrict__ out) {
    __shared__ float sd[DH];
    const int i = blockIdx.x, t = threadIdx.x;               // blockDim = 64
    const float v = x[(size_t)i * DH + t];
    sd[t] = v * v; __syncthreads();
    for (int st = 32; st > 0; st >>= 1) {
        if (t < st) sd[t] += sd[t + st];
        __syncthreads();
    }
    out[(size_t)i * DH + t] = (_Float16)(v * rsqrtf(sd[0]));
}

// ------- per-tile contribution: exp + row/col sums + pos-masked sums -------
// C/D layout: element v of lane (l15,grp) is sim[rbase + v + 8*grp][cbase + l15].
// Psc = pos tile (rbase.., cbase..) row-major; Pmp = pos tile (cbase.., rbase..).
__device__ __forceinline__ void heco_tile_contrib(
    const v8f& c, const float* Psc, const float* Pmp,
    int rbase, int cbase, int l15, int grp,
    float* rowsum, float* colsum, float* pd_sc, float* pd_mp) {
    float cs = 0.f, cm = 0.f;
#pragma unroll
    for (int v = 0; v < 8; ++v) {
        const int moff = v + grp * 8;
        const float e  = __expf(c[v] * INV_TAU);
        const float pv = Psc[moff * 16 + l15];   // pos[r, c]
        const float pt = Pmp[l15 * 16 + moff];   // pos[c, r] (from LDS, no strided HBM)
        // row reductions stay inside one 16-lane group (xor strides < 16)
        float rs = e, rp = e * pv;
#pragma unroll
        for (int s = 1; s < 16; s <<= 1) {
            rs += __shfl_xor(rs, s, 32);
            rp += __shfl_xor(rp, s, 32);
        }
        if (l15 == 0) {
            atomicAdd(&rowsum[rbase + moff], rs);
            atomicAdd(&pd_sc[rbase + moff], rp);
        }
        cs += e;
        cm += e * pt;
    }
    cs += __shfl_xor(cs, 16, 32);                 // merge the two M-halves
    cm += __shfl_xor(cm, 16, 32);
    if (grp == 0) {
        atomicAdd(&colsum[cbase + l15], cs);
        atomicAdd(&pd_mp[cbase + l15], cm);
    }
}

// ------- fused sim GEMM over symmetric tile pairs --------------------------
// Wave handles pair {(tm,tn),(tn,tm)} for tm <= tn: 4 WMMAs, both pos tiles
// staged coalesced into LDS -> pos streamed from HBM exactly once.
__global__ void heco_sim_pair(const _Float16* __restrict__ zs,
                              const _Float16* __restrict__ zm,
                              const float* __restrict__ pos,
                              float* __restrict__ rowsum, float* __restrict__ colsum,
                              float* __restrict__ pd_sc,  float* __restrict__ pd_mp) {
    __shared__ float4 pshare[4 * 128];             // 4 waves x 2 tiles x 1KB
    const int wave = threadIdx.x >> 5, lane = threadIdx.x & 31;
    const int tm = (blockIdx.y * 4 + wave) << 4;   // gridDim.y = 128
    const int tn = blockIdx.x << 4;                // gridDim.x = 512
    const bool active = (tm <= tn);                // wave-uniform predicate
    const bool diag   = (tm == tn);
    const int l15 = lane & 15, grp = lane >> 4;

    float* P1 = (float*)(pshare + wave * 128);     // pos tile (tm.., tn..)
    float* P2 = diag ? P1 : P1 + 256;              // pos tile (tn.., tm..)

    v8f c1 = {}, c2 = {};
    if (active) {
        // ---- sim tile (tm,tn): zs rows tm x zm rows tn ----
        const _Float16* a1 = zs + (size_t)(tm + l15) * DH + grp * 8;
        const _Float16* b1 = zm + (size_t)(tn + l15) * DH + grp * 8;
#pragma unroll
        for (int k0 = 0; k0 < DH; k0 += 32) {
            union { v16h v; uint4 q[2]; } a, b;
            a.q[0] = *(const uint4*)(a1 + k0);
            a.q[1] = *(const uint4*)(a1 + k0 + 16);
            b.q[0] = *(const uint4*)(b1 + k0);
            b.q[1] = *(const uint4*)(b1 + k0 + 16);
            c1 = __builtin_amdgcn_wmma_f32_16x16x32_f16(
                     false, a.v, false, b.v, (short)0, c1, false, false);
        }
        if (!diag) {
            // ---- mirror tile (tn,tm): zs rows tn x zm rows tm ----
            const _Float16* a2 = zs + (size_t)(tn + l15) * DH + grp * 8;
            const _Float16* b2 = zm + (size_t)(tm + l15) * DH + grp * 8;
#pragma unroll
            for (int k0 = 0; k0 < DH; k0 += 32) {
                union { v16h v; uint4 q[2]; } a, b;
                a.q[0] = *(const uint4*)(a2 + k0);
                a.q[1] = *(const uint4*)(a2 + k0 + 16);
                b.q[0] = *(const uint4*)(b2 + k0);
                b.q[1] = *(const uint4*)(b2 + k0 + 16);
                c2 = __builtin_amdgcn_wmma_f32_16x16x32_f16(
                         false, a.v, false, b.v, (short)0, c2, false, false);
            }
        }
        // ---- stage pos tiles into LDS, fully coalesced (64B per lane) ----
        const int q0 = lane, q1 = lane + 32;       // 64 float4 per 16x16 tile
        ((float4*)P1)[q0] = *(const float4*)(pos + (size_t)(tm + (q0 >> 2)) * NN + tn + ((q0 & 3) << 2));
        ((float4*)P1)[q1] = *(const float4*)(pos + (size_t)(tm + (q1 >> 2)) * NN + tn + ((q1 & 3) << 2));
        if (!diag) {
            ((float4*)P2)[q0] = *(const float4*)(pos + (size_t)(tn + (q0 >> 2)) * NN + tm + ((q0 & 3) << 2));
            ((float4*)P2)[q1] = *(const float4*)(pos + (size_t)(tn + (q1 >> 2)) * NN + tm + ((q1 & 3) << 2));
        }
    }
    __syncthreads();                               // all waves reach (no early return)
    if (active) {
        heco_tile_contrib(c1, P1, P2, tm, tn, l15, grp, rowsum, colsum, pd_sc, pd_mp);
        if (!diag)
            heco_tile_contrib(c2, P2, P1, tn, tm, l15, grp, rowsum, colsum, pd_sc, pd_mp);
    }
}

// ---------------- final loss reduction -------------------------------------
__global__ void heco_loss(const float* __restrict__ rowsum,
                          const float* __restrict__ colsum,
                          const float* __restrict__ pd_sc,
                          const float* __restrict__ pd_mp,
                          float* __restrict__ out) {
    __shared__ float sd[256];
    const int i = blockIdx.x * 256 + threadIdx.x;
    float t = 0.f;
    if (i < NN) {
        const float lsc = logf(rowsum[i] + 1e-8f) - logf(pd_sc[i]);
        const float lmp = logf(colsum[i] + 1e-8f) - logf(pd_mp[i]);
        t = 0.5f * lsc + 0.5f * lmp;              // LAMBDA = 0.5
    }
    sd[threadIdx.x] = t; __syncthreads();
    for (int st = 128; st > 0; st >>= 1) {
        if (threadIdx.x < st) sd[threadIdx.x] += sd[threadIdx.x + st];
        __syncthreads();
    }
    if (threadIdx.x == 0) atomicAdd(out, sd[0] * (1.0f / NN));
}

// ---------------------------------------------------------------------------
extern "C" void kernel_launch(void* const* d_in, const int* in_sizes, int n_in,
                              void* d_out, int out_size, void* d_ws, size_t ws_size,
                              hipStream_t stream) {
    // ---- input mapping (setup_inputs insertion order, lists expanded) ----
    const float* feat0 = (const float*)d_in[0];
    const float* feat1 = (const float*)d_in[1];
    const float* feat2 = (const float*)d_in[2];
    const float* pos   = (const float*)d_in[3];
    const int* src_b1  = (const int*)d_in[4];
    const int* src_b2  = (const int*)d_in[6];
    const int* src_m1  = (const int*)d_in[8];
    const int* src_m2  = (const int*)d_in[10];
    const float* fc_w[3]  = {(const float*)d_in[12], (const float*)d_in[13], (const float*)d_in[14]};
    const float* fc_b[3]  = {(const float*)d_in[15], (const float*)d_in[16], (const float*)d_in[17]};
    const float* attn_l[2] = {(const float*)d_in[18], (const float*)d_in[19]};
    const float* attn_r[2] = {(const float*)d_in[20], (const float*)d_in[21]};
    const float* gcn_w[2] = {(const float*)d_in[22], (const float*)d_in[23]};
    const float* gcn_b[2] = {(const float*)d_in[24], (const float*)d_in[25]};
    const float* prelu[2] = {(const float*)d_in[26], (const float*)d_in[27]};
    const float* sc_w = (const float*)d_in[28];
    const float* sc_b = (const float*)d_in[29];
    const float* sc_v = (const float*)d_in[30];
    const float* mp_w = (const float*)d_in[31];
    const float* mp_b = (const float*)d_in[32];
    const float* mp_v = (const float*)d_in[33];
    const float* pw1  = (const float*)d_in[34];
    const float* pb1  = (const float*)d_in[35];
    const float* pw2  = (const float*)d_in[36];
    const float* pb2  = (const float*)d_in[37];
    float* out = (float*)d_out;

    // ---- workspace layout (256B aligned slices) ----
    char* ws = (char*)d_ws; size_t off = 0;
    auto alloc = [&](size_t bytes) -> void* {
        void* p = ws + off; off = (off + bytes + 255) & ~(size_t)255; return p;
    };
    _Float16* f0h = (_Float16*)alloc((size_t)NN   * DIN * 2);
    _Float16* f1h = (_Float16*)alloc((size_t)NNBR * DIN * 2);
    _Float16* f2h = (_Float16*)alloc((size_t)NNBR * DIN * 2);
    _Float16* h0h = (_Float16*)alloc((size_t)NN   * DH * 2);
    _Float16* h1h = (_Float16*)alloc((size_t)NNBR * DH * 2);
    _Float16* h2h = (_Float16*)alloc((size_t)NNBR * DH * 2);
    _Float16* g1h = (_Float16*)alloc((size_t)NN * DH * 2);
    _Float16* g2h = (_Float16*)alloc((size_t)NN * DH * 2);
    _Float16* xw1 = (_Float16*)alloc((size_t)NN * DH * 2);
    _Float16* xw2 = (_Float16*)alloc((size_t)NN * DH * 2);
    _Float16* m1h = (_Float16*)alloc((size_t)NN * DH * 2);
    _Float16* m2h = (_Float16*)alloc((size_t)NN * DH * 2);
    _Float16* zsc = (_Float16*)alloc((size_t)NN * DH * 2);
    _Float16* zmp = (_Float16*)alloc((size_t)NN * DH * 2);
    _Float16* thb = (_Float16*)alloc((size_t)NN * DH * 2);   // tanh scratch
    _Float16* p1h = (_Float16*)alloc((size_t)NN * DH * 2);   // proj hidden
    float*    pf  = (float*)   alloc((size_t)NN * DH * 4);   // proj out (f32)
    _Float16* zs_h = (_Float16*)alloc((size_t)NN * DH * 2);
    _Float16* zm_h = (_Float16*)alloc((size_t)NN * DH * 2);
    // f16 weights
    _Float16* fcw_h[3]; for (int i = 0; i < 3; ++i) fcw_h[i] = (_Float16*)alloc(DH * DIN * 2);
    _Float16* gcnw_h[2]; for (int i = 0; i < 2; ++i) gcnw_h[i] = (_Float16*)alloc(DH * DH * 2);
    _Float16* scw_h = (_Float16*)alloc(DH * DH * 2);
    _Float16* mpw_h = (_Float16*)alloc(DH * DH * 2);
    _Float16* pw1_h = (_Float16*)alloc(DH * DH * 2);
    _Float16* pw2_h = (_Float16*)alloc(DH * DH * 2);
    // accumulators (contiguous -> single memset)
    float* rowsum = (float*)alloc(NN * 4 * 4);
    float* colsum = rowsum + NN;
    float* pd_sc  = rowsum + 2 * NN;
    float* pd_mp  = rowsum + 3 * NN;
    float* wbuf   = (float*)alloc(8 * 4);   // [w_sc0,w_sc1,w_mp0,w_mp1]
    float* betab  = wbuf + 4;               // [b_sc0,b_sc1,b_mp0,b_mp1]

    // ---- zero accumulators + output ----
    hipMemsetAsync(rowsum, 0, (size_t)NN * 4 * 4, stream);
    hipMemsetAsync(wbuf, 0, 4 * 4, stream);
    hipMemsetAsync(out, 0, 4, stream);

    // ---- f32 -> f16 conversions ----
    heco_cvt_f16<<<2048, 256, 0, stream>>>(feat0, f0h, NN * DIN);
    heco_cvt_f16<<<4096, 256, 0, stream>>>(feat1, f1h, NNBR * DIN);
    heco_cvt_f16<<<4096, 256, 0, stream>>>(feat2, f2h, NNBR * DIN);
    for (int i = 0; i < 3; ++i) heco_cvt_f16<<<32, 256, 0, stream>>>(fc_w[i], fcw_h[i], DH * DIN);
    for (int i = 0; i < 2; ++i) heco_cvt_f16<<<16, 256, 0, stream>>>(gcn_w[i], gcnw_h[i], DH * DH);
    heco_cvt_f16<<<16, 256, 0, stream>>>(sc_w, scw_h, DH * DH);
    heco_cvt_f16<<<16, 256, 0, stream>>>(mp_w, mpw_h, DH * DH);
    heco_cvt_f16<<<16, 256, 0, stream>>>(pw1, pw1_h, DH * DH);
    heco_cvt_f16<<<16, 256, 0, stream>>>(pw2, pw2_h, DH * DH);

    auto linear = [&](const _Float16* X, const _Float16* W, const float* b,
                      _Float16* Yh, float* Yf, int M, int N, int K, int act) {
        int tiles = (M / 16) * (N / 16);
        int blocks = (tiles + 3) / 4;       // 4 waves (16x16 tiles) per block
        heco_wmma_linear<<<blocks, 128, 0, stream>>>(X, W, b, Yh, Yf, M, N, K, act);
    };

    // ---- feature projections: h = elu(feat @ W^T + b) ----
    linear(f0h, fcw_h[0], fc_b[0], h0h, nullptr, NN,   DH, DIN, 1);
    linear(f1h, fcw_h[1], fc_b[1], h1h, nullptr, NNBR, DH, DIN, 1);
    linear(f2h, fcw_h[2], fc_b[2], h2h, nullptr, NNBR, DH, DIN, 1);

    // ---- network-schema view: two GATs ----
    heco_gat<<<NN, 64, 0, stream>>>(h1h, h0h, src_b1, attn_l[0], attn_r[0], g1h);
    heco_gat<<<NN, 64, 0, stream>>>(h2h, h0h, src_b2, attn_l[1], attn_r[1], g2h);

    // ---- metapath view: two GCNs (xW then deg-16 mean + bias + PReLU) ----
    linear(h0h, gcnw_h[0], nullptr, xw1, nullptr, NN, DH, DH, 0);
    linear(h0h, gcnw_h[1], nullptr, xw2, nullptr, NN, DH, DH, 0);
    heco_gcn_agg<<<NN, 64, 0, stream>>>(xw1, src_m1, gcn_b[0], prelu[0], m1h);
    heco_gcn_agg<<<NN, 64, 0, stream>>>(xw2, src_m2, gcn_b[1], prelu[1], m2h);

    // ---- semantic attention (sc view) ----
    linear(g1h, scw_h, sc_b, thb, nullptr, NN, DH, DH, 2);
    heco_semred<<<256, 256, 0, stream>>>(thb, sc_v, wbuf + 0);
    linear(g2h, scw_h, sc_b, thb, nullptr, NN, DH, DH, 2);
    heco_semred<<<256, 256, 0, stream>>>(thb, sc_v, wbuf + 1);
    heco_softmax2<<<1, 1, 0, stream>>>(wbuf + 0, betab + 0);
    heco_combine2<<<256, 256, 0, stream>>>(g1h, g2h, betab + 0, zsc, NN * DH);

    // ---- semantic attention (mp view) ----
    linear(m1h, mpw_h, mp_b, thb, nullptr, NN, DH, DH, 2);
    heco_semred<<<256, 256, 0, stream>>>(thb, mp_v, wbuf + 2);
    linear(m2h, mpw_h, mp_b, thb, nullptr, NN, DH, DH, 2);
    heco_semred<<<256, 256, 0, stream>>>(thb, mp_v, wbuf + 3);
    heco_softmax2<<<1, 1, 0, stream>>>(wbuf + 2, betab + 2);
    heco_combine2<<<256, 256, 0, stream>>>(m1h, m2h, betab + 2, zmp, NN * DH);

    // ---- contrast: shared projection MLP + L2 normalize ----
    linear(zsc, pw1_h, pb1, p1h, nullptr, NN, DH, DH, 1);
    linear(p1h, pw2_h, pb2, nullptr, pf, NN, DH, DH, 0);
    heco_l2norm<<<NN, 64, 0, stream>>>(pf, zs_h);
    linear(zmp, pw1_h, pb1, p1h, nullptr, NN, DH, DH, 1);
    linear(p1h, pw2_h, pb2, nullptr, pf, NN, DH, DH, 0);
    heco_l2norm<<<NN, 64, 0, stream>>>(pf, zm_h);

    // ---- fused sim GEMM over symmetric tile pairs (pos read once) ----
    heco_sim_pair<<<dim3(NN / 16, NN / 64), 128, 0, stream>>>(
        zs_h, zm_h, pos, rowsum, colsum, pd_sc, pd_mp);

    // ---- final loss ----
    heco_loss<<<NN / 256, 256, 0, stream>>>(rowsum, colsum, pd_sc, pd_mp, out);
}
